// BitLinear_75041668596052
// MI455X (gfx1250) — 8-bit matrix operand order repaired and checked by reading, compiled, not run on hardware
//
#include <hip/hip_runtime.h>
#include <hip/hip_bf16.h>

typedef int   v8i  __attribute__((ext_vector_type(8)));
typedef int   v16i __attribute__((ext_vector_type(16)));
typedef float v8f  __attribute__((ext_vector_type(8)));

#define DIM 2048
#define HID 8192

__device__ __forceinline__ unsigned lds_off_u32(const void* p) {
  return (unsigned)(unsigned long long)p;
}
__device__ __forceinline__ void async_copy_b128(unsigned lds_dst,
                                                const void* gsrc) {
  asm volatile("global_load_async_to_lds_b128 %0, %1, off"
               :: "v"(lds_dst), "v"(gsrc) : "memory");
}
__device__ __forceinline__ void async_copy_b128_off16(unsigned lds_dst,
                                                      const void* gsrc) {
  asm volatile("global_load_async_to_lds_b128 %0, %1, off offset:16"
               :: "v"(lds_dst), "v"(gsrc) : "memory");
}
__device__ __forceinline__ void async_copy_b128_off32(unsigned lds_dst,
                                                      const void* gsrc) {
  asm volatile("global_load_async_to_lds_b128 %0, %1, off offset:32"
               :: "v"(lds_dst), "v"(gsrc) : "memory");
}
__device__ __forceinline__ void async_copy_b128_off48(unsigned lds_dst,
                                                      const void* gsrc) {
  asm volatile("global_load_async_to_lds_b128 %0, %1, off offset:48"
               :: "v"(lds_dst), "v"(gsrc) : "memory");
}
__device__ __forceinline__ void wait_async0() {
  asm volatile("s_wait_asynccnt 0x0" ::: "memory");
}

__device__ __forceinline__ unsigned char f32_to_e4m3(float x) {
  unsigned int ux   = __float_as_uint(x);
  unsigned int sign = (ux >> 24) & 0x80u;
  float ax = __uint_as_float(ux & 0x7FFFFFFFu);
  ax = fminf(ax, 448.0f);
  if (ax < 0.0078125f) {
    int q = (int)rintf(ax * 512.0f);
    return (unsigned char)(sign | (unsigned)q);
  }
  unsigned int u = __float_as_uint(ax);
  int e = (int)((u >> 23) & 0xFFu) - 127;
  unsigned int m    = u & 0x7FFFFFu;
  unsigned int mant = m >> 20;
  unsigned int rem  = m & 0xFFFFFu;
  if (rem > 0x80000u || (rem == 0x80000u && (mant & 1u))) {
    mant++;
    if (mant == 8u) { mant = 0u; e++; }
  }
  if (e > 8) { e = 8; mant = 6u; }
  if (e < -6) {
    int q = (int)rintf(ax * 512.0f);
    if (q >= 8) return (unsigned char)(sign | 0x08u);
    return (unsigned char)(sign | (unsigned)q);
  }
  return (unsigned char)(sign | ((unsigned)(e + 7) << 3) | mant);
}

__global__ __launch_bounds__(256) void bl_abssum_partial(
    const float* __restrict__ w, long n, float* __restrict__ part) {
  __shared__ float red[256];
  const int tid = threadIdx.x;
  float s = 0.0f;
  for (long i = (long)blockIdx.x * 256 + tid; i < n; i += (long)gridDim.x * 256)
    s += fabsf(w[i]);
  red[tid] = s; __syncthreads();
  for (int st = 128; st > 0; st >>= 1) {
    if (tid < st) red[tid] += red[tid + st];
    __syncthreads();
  }
  if (tid == 0) part[blockIdx.x] = red[0];
}

__global__ __launch_bounds__(256) void bl_finalize_scales(
    const float* __restrict__ pu, const float* __restrict__ pd, int nparts,
    float* __restrict__ scales, float inv_n) {
  __shared__ float red[256];
  const int tid = threadIdx.x;
  float s = 0.0f;
  for (int i = tid; i < nparts; i += 256) s += pu[i];
  red[tid] = s; __syncthreads();
  for (int st = 128; st > 0; st >>= 1) {
    if (tid < st) red[tid] += red[tid + st];
    __syncthreads();
  }
  float su = fmaxf(red[0] * inv_n, 1e-8f);
  __syncthreads();
  s = 0.0f;
  for (int i = tid; i < nparts; i += 256) s += pd[i];
  red[tid] = s; __syncthreads();
  for (int st = 128; st > 0; st >>= 1) {
    if (tid < st) red[tid] += red[tid + st];
    __syncthreads();
  }
  if (tid == 0) {
    float sd = fmaxf(red[0] * inv_n, 1e-8f);
    scales[0] = su; scales[1] = sd;
    scales[2] = 1.0f / su; scales[3] = 1.0f / sd;
  }
}

__global__ __launch_bounds__(256) void bl_quant_up(
    const float* __restrict__ w, signed char* __restrict__ wq,
    const float* __restrict__ scales, long n) {
  long i = ((long)blockIdx.x * 256 + threadIdx.x) * 4;
  if (i >= n) return;
  const float inv = scales[2];
  float4 v = *(const float4*)(w + i);
  char4 o;
  o.x = (signed char)(int)fminf(fmaxf(rintf(v.x * inv), -1.0f), 1.0f);
  o.y = (signed char)(int)fminf(fmaxf(rintf(v.y * inv), -1.0f), 1.0f);
  o.z = (signed char)(int)fminf(fmaxf(rintf(v.z * inv), -1.0f), 1.0f);
  o.w = (signed char)(int)fminf(fmaxf(rintf(v.w * inv), -1.0f), 1.0f);
  *(char4*)(wq + i) = o;
}

__global__ __launch_bounds__(256) void bl_quant_dn(
    const float* __restrict__ w, unsigned char* __restrict__ wq,
    const float* __restrict__ scales, long n) {
  long i = ((long)blockIdx.x * 256 + threadIdx.x) * 4;
  if (i >= n) return;
  const float inv = scales[3];
  float4 v = *(const float4*)(w + i);
  int qx = (int)fminf(fmaxf(rintf(v.x * inv), -1.0f), 1.0f);
  int qy = (int)fminf(fmaxf(rintf(v.y * inv), -1.0f), 1.0f);
  int qz = (int)fminf(fmaxf(rintf(v.z * inv), -1.0f), 1.0f);
  int qw = (int)fminf(fmaxf(rintf(v.w * inv), -1.0f), 1.0f);
  uchar4 o;
  o.x = qx == 0 ? 0x00u : (qx > 0 ? 0x38u : 0xB8u);
  o.y = qy == 0 ? 0x00u : (qy > 0 ? 0x38u : 0xB8u);
  o.z = qz == 0 ? 0x00u : (qz > 0 ? 0x38u : 0xB8u);
  o.w = qw == 0 ? 0x00u : (qw > 0 ? 0x38u : 0xB8u);
  *(uchar4*)(wq + i) = o;
}

__global__ __launch_bounds__(256) void bl_act_quant(
    const float* __restrict__ x, const float* __restrict__ nw,
    signed char* __restrict__ xq, float* __restrict__ sx) {
  __shared__ float red[256];
  const int t = blockIdx.x, tid = threadIdx.x;
  const float* row = x + (long)t * DIM;
  float v[8];
  float ss = 0.0f;
#pragma unroll
  for (int i = 0; i < 8; i++) { v[i] = row[tid + i * 256]; ss += v[i] * v[i]; }
  red[tid] = ss; __syncthreads();
  for (int st = 128; st > 0; st >>= 1) {
    if (tid < st) red[tid] += red[tid + st];
    __syncthreads();
  }
  const float rstd = rsqrtf(red[0] * (1.0f / (float)DIM) + 1e-6f);
  __syncthreads();
  float amax = 0.0f;
#pragma unroll
  for (int i = 0; i < 8; i++) {
    v[i] = v[i] * rstd * nw[tid + i * 256];
    amax = fmaxf(amax, fabsf(v[i]));
  }
  red[tid] = amax; __syncthreads();
  for (int st = 128; st > 0; st >>= 1) {
    if (tid < st) red[tid] = fmaxf(red[tid], red[tid + st]);
    __syncthreads();
  }
  const float g = fmaxf(red[0], 1e-8f);
  if (tid == 0) sx[t] = g * (1.0f / 127.0f);
  const float q = 127.0f / g;
#pragma unroll
  for (int i = 0; i < 8; i++) {
    float r = fminf(fmaxf(rintf(v[i] * q), -127.0f), 127.0f);
    xq[(long)t * DIM + tid + i * 256] = (signed char)(int)r;
  }
}

__global__ __launch_bounds__(256) void bl_gemm1_iu8(
    const signed char* __restrict__ xq, const signed char* __restrict__ wq_up,
    const float* __restrict__ sx, const float* __restrict__ scales,
    unsigned char* __restrict__ hq) {
  alignas(16) __shared__ signed char lA[128 * 64];
  const int tid  = threadIdx.x;
  const int lane = tid & 31;
  const int wave = tid >> 5;
  const int n0 = blockIdx.x * 128;
  const int t0 = blockIdx.y * 128;
  const int wm = (wave >> 1) * 32;
  const int wn = (wave & 1) * 64;
  const int lrow = lane & 15;
  const int lhi  = lane >> 4;
  const float s_up = scales[0];

  const int sr = tid >> 1, soff = (tid & 1) * 32;
  const unsigned lds_dst = lds_off_u32(lA + sr * 64 + soff);

  v8i acc[2][4];
  const v8i z8 = {0, 0, 0, 0, 0, 0, 0, 0};
#pragma unroll
  for (int mi = 0; mi < 2; mi++)
#pragma unroll
    for (int ni = 0; ni < 4; ni++) acc[mi][ni] = z8;

  for (int k0 = 0; k0 < DIM; k0 += 64) {
    {
      const signed char* gsrc = xq + (long)(t0 + sr) * DIM + k0 + soff;
      async_copy_b128(lds_dst, gsrc);
      async_copy_b128_off16(lds_dst, gsrc);
      wait_async0();
    }
    __syncthreads();
    v8i af[2];
#pragma unroll
    for (int mi = 0; mi < 2; mi++) {
      const signed char* base = lA + (wm + mi * 16 + lrow) * 64 + lhi * 16;
      int2 p0 = *(const int2*)(base);
      int2 p1 = *(const int2*)(base + 8);
      int2 p2 = *(const int2*)(base + 32);
      int2 p3 = *(const int2*)(base + 40);
      v8i a = {p0.x, p0.y, p1.x, p1.y, p2.x, p2.y, p3.x, p3.y};
      af[mi] = a;
    }
#pragma unroll
    for (int ni = 0; ni < 4; ni++) {
      const signed char* bb =
          wq_up + (long)(n0 + wn + ni * 16 + lrow) * DIM + k0 + lhi * 16;
      int4 q0 = *(const int4*)(bb);
      int4 q1 = *(const int4*)(bb + 32);
      v8i bf = {q0.x, q0.y, q0.z, q0.w, q1.x, q1.y, q1.z, q1.w};
      acc[0][ni] = __builtin_amdgcn_wmma_i32_16x16x64_iu8(
          true, af[0], true, bf, acc[0][ni], false, false);
      acc[1][ni] = __builtin_amdgcn_wmma_i32_16x16x64_iu8(
          true, af[1], true, bf, acc[1][ni], false, false);
    }
    __syncthreads();
  }
#pragma unroll
  for (int mi = 0; mi < 2; mi++) {
#pragma unroll
    for (int j = 0; j < 8; j++) {
      const int t = t0 + wm + mi * 16 + lhi * 8 + j;
      const float sv = sx[t] * s_up;
#pragma unroll
      for (int ni = 0; ni < 4; ni++) {
        float f = (float)acc[mi][ni][j] * sv;
        f = f / (1.0f + __expf(-f));
        hq[(long)t * HID + n0 + wn + ni * 16 + lrow] = f32_to_e4m3(f * 16.0f);
      }
    }
  }
}

__global__ __launch_bounds__(256) void bl_gemm2_fp8(
    const unsigned char* __restrict__ hq, const unsigned char* __restrict__ wq_dn,
    const float* __restrict__ x, const float* __restrict__ gamma,
    const float* __restrict__ scales, float* __restrict__ out) {
  alignas(16) __shared__ unsigned char lA[128 * 128];
  const int tid  = threadIdx.x;
  const int lane = tid & 31;
  const int wave = tid >> 5;
  const int n0 = blockIdx.x * 128;
  const int t0 = blockIdx.y * 128;
  const int wm = (wave >> 1) * 32;
  const int wn = (wave & 1) * 64;
  const int lrow = lane & 15;
  const int lhi  = lane >> 4;
  const float sd = scales[1] * (1.0f / 16.0f);

  const int sr = tid >> 1, soff = (tid & 1) * 64;
  const unsigned lds_dst = lds_off_u32(lA + sr * 128 + soff);

  v8f acc[2][4];
  const v8f zf = {0.f, 0.f, 0.f, 0.f, 0.f, 0.f, 0.f, 0.f};
#pragma unroll
  for (int mi = 0; mi < 2; mi++)
#pragma unroll
    for (int ni = 0; ni < 4; ni++) acc[mi][ni] = zf;

  for (int k0 = 0; k0 < HID; k0 += 128) {
    {
      const unsigned char* gsrc = hq + (long)(t0 + sr) * HID + k0 + soff;
      async_copy_b128(lds_dst, gsrc);
      async_copy_b128_off16(lds_dst, gsrc);
      async_copy_b128_off32(lds_dst, gsrc);
      async_copy_b128_off48(lds_dst, gsrc);
      wait_async0();
    }
    __syncthreads();
    v16i af[2];
#pragma unroll
    for (int mi = 0; mi < 2; mi++) {
      const unsigned char* base = lA + (wm + mi * 16 + lrow) * 128 + lhi * 16;
      int2 p0 = *(const int2*)(base);
      int2 p1 = *(const int2*)(base + 8);
      int2 p2 = *(const int2*)(base + 32);
      int2 p3 = *(const int2*)(base + 40);
      int2 p4 = *(const int2*)(base + 64);
      int2 p5 = *(const int2*)(base + 72);
      int2 p6 = *(const int2*)(base + 96);
      int2 p7 = *(const int2*)(base + 104);
      v16i a = {p0.x, p0.y, p1.x, p1.y, p2.x, p2.y, p3.x, p3.y,
                p4.x, p4.y, p5.x, p5.y, p6.x, p6.y, p7.x, p7.y};
      af[mi] = a;
    }
#pragma unroll
    for (int ni = 0; ni < 4; ni++) {
      const unsigned char* bb =
          wq_dn + (long)(n0 + wn + ni * 16 + lrow) * HID + k0 + lhi * 16;
      int4 q0 = *(const int4*)(bb);
      int4 q1 = *(const int4*)(bb + 32);
      int4 q2 = *(const int4*)(bb + 64);
      int4 q3 = *(const int4*)(bb + 96);
      v16i bf = {q0.x, q0.y, q0.z, q0.w, q1.x, q1.y, q1.z, q1.w,
                 q2.x, q2.y, q2.z, q2.w, q3.x, q3.y, q3.z, q3.w};
      acc[0][ni] = __builtin_amdgcn_wmma_f32_16x16x128_fp8_fp8(
          af[0], bf, (short)0, acc[0][ni], false, false);
      acc[1][ni] = __builtin_amdgcn_wmma_f32_16x16x128_fp8_fp8(
          af[1], bf, (short)0, acc[1][ni], false, false);
    }
    __syncthreads();
  }
#pragma unroll
  for (int mi = 0; mi < 2; mi++) {
#pragma unroll
    for (int j = 0; j < 8; j++) {
      const int t = t0 + wm + mi * 16 + lhi * 8 + j;
#pragma unroll
      for (int ni = 0; ni < 4; ni++) {
        const int c = n0 + wn + ni * 16 + lrow;
        const long idx = (long)t * DIM + c;
        out[idx] = x[idx] + gamma[c] * sd * acc[mi][ni][j];
      }
    }
  }
}

extern "C" void kernel_launch(void* const* d_in, const int* in_sizes, int n_in,
                              void* d_out, int out_size, void* d_ws, size_t ws_size,
                              hipStream_t stream) {
  const float* x  = (const float*)d_in[0];
  const float* wu = (const float*)d_in[1];
  const float* wd = (const float*)d_in[2];
  const float* nw = (const float*)d_in[3];
  const float* gm = (const float*)d_in[4];
  float* out = (float*)d_out;

  const int  T    = in_sizes[0] / DIM;
  const long n_up = (long)HID * DIM;
  const long n_dn = (long)DIM * HID;

  char* ws = (char*)d_ws;
  float*         part_up = (float*)(ws);
  float*         part_dn = (float*)(ws + 8192);
  float*         scales  = (float*)(ws + 16384);
  float*         sx      = (float*)(ws + 16640);
  signed char*   wq_up   = (signed char*)(ws + 82176);
  unsigned char* wq_dn   = (unsigned char*)(ws + 82176 + 16777216L);
  signed char*   xq      = (signed char*)(ws + 82176 + 2 * 16777216L);
  unsigned char* hq      = (unsigned char*)(ws + 82176 + 2 * 16777216L + 33554432L);

  bl_abssum_partial<<<2048, 256, 0, stream>>>(wu, n_up, part_up);
  bl_abssum_partial<<<2048, 256, 0, stream>>>(wd, n_dn, part_dn);
  bl_finalize_scales<<<1, 256, 0, stream>>>(part_up, part_dn, 2048, scales,
                                            1.0f / (float)n_up);
  bl_quant_up<<<(int)(n_up / 1024), 256, 0, stream>>>(wu, wq_up, scales, n_up);
  bl_quant_dn<<<(int)(n_dn / 1024), 256, 0, stream>>>(wd, wq_dn, scales, n_dn);
  bl_act_quant<<<T, 256, 0, stream>>>(x, nw, xq, sx);
  bl_gemm1_iu8<<<dim3(HID / 128, T / 128), 256, 0, stream>>>(xq, wq_up, sx,
                                                             scales, hq);
  bl_gemm2_fp8<<<dim3(DIM / 128, T / 128), 256, 0, stream>>>(hq, wq_dn, x, gm,
                                                             scales, out);
}
